// DiffMultiHeadedAttention_8280696947243
// MI455X (gfx1250) — matrix operand order repaired and checked by reading, not compiled, not run on hardware
//
#include <hip/hip_runtime.h>
#include <hip/hip_bf16.h>

// ---------------------------------------------------------------------------
// Types for CDNA5 WMMA (wave32): V_WMMA_F32_16X16X32_BF16
// ---------------------------------------------------------------------------
typedef __attribute__((ext_vector_type(16))) __bf16 v16bf;
typedef __attribute__((ext_vector_type(8)))  __bf16 v8bf;
typedef __attribute__((ext_vector_type(8)))  float  v8f;
typedef int v4i __attribute__((vector_size(16)));   // async-LDS builtin elem type

#define DEV __device__ __forceinline__
#define AS1 __attribute__((address_space(1)))
#define AS3 __attribute__((address_space(3)))

// CDNA5 async VMEM->LDS path (ASYNCcnt-tracked)
#if defined(__has_builtin)
#if __has_builtin(__builtin_amdgcn_global_load_async_to_lds_b128)
#define HAVE_ASYNC_LDS 1
#endif
#endif

// Problem constants (from reference)
constexpr int   BBATCH = 2;
constexpr int   TDIM   = 2048;
constexpr int   NPOS   = 2048;
constexpr int   HID    = 1024;
constexpr int   NHEAD  = 16;
constexpr int   DH     = 64;
constexpr int   HALF   = 32;
constexpr float LAMBDA_INIT = 0.7008206670670481f;   // 0.8 - 0.6*exp(-1.8)
constexpr float QK_SCALE    = 0.17677669529663687f;  // 1/sqrt(32)

DEV __bf16 f2bf(float f) {
  unsigned u = __builtin_bit_cast(unsigned, f);
  unsigned short r = (unsigned short)((u + 0x7FFFu + ((u >> 16) & 1u)) >> 16);
  return __builtin_bit_cast(__bf16, r);
}

DEV v16bf cat16(v8bf lo, v8bf hi) {
  v16bf r;
#pragma unroll
  for (int e = 0; e < 8; ++e) { r[e] = lo[e]; r[e + 8] = hi[e]; }
  return r;
}

// A-fragment (16x32 bf16, row-major source, row stride `ld`):
// lane: M = lane%16 ; K = (lane<16?0:16)+e (16 contiguous)
DEV v16bf load_a_frag(const __bf16* __restrict__ base, int ld, int l16, int hi) {
  const __bf16* p = base + (size_t)l16 * ld + (hi ? 16 : 0);
  v8bf lo = *reinterpret_cast<const v8bf*>(p);
  v8bf hb = *reinterpret_cast<const v8bf*>(p + 8);
  return cat16(lo, hb);
}

// B-fragment (32x16 bf16): lane: N = lane%16, K = (lane<16?0:16)+e -> 16
// contiguous bf16 from the (transposed-operand) source row.
DEV v16bf load_b_frag(const __bf16* __restrict__ base, int ld, int l16, int hi) {
  const __bf16* p = base + (size_t)l16 * ld + (hi ? 16 : 0);
  v8bf lo = *reinterpret_cast<const v8bf*>(p);
  v8bf hb = *reinterpret_cast<const v8bf*>(p + 8);
  return cat16(lo, hb);
}

// 16-byte global -> LDS copy: async on CDNA5 when available
DEV void async_cp16(const __bf16* g, __bf16* l) {
#ifdef HAVE_ASYNC_LDS
  __builtin_amdgcn_global_load_async_to_lds_b128((AS1 v4i*)g, (AS3 v4i*)l, 0, 0);
#else
  *reinterpret_cast<v8bf*>(l) = *reinterpret_cast<const v8bf*>(g);
#endif
}

DEV void wait_async_lds() {
#ifdef HAVE_ASYNC_LDS
#if __has_builtin(__builtin_amdgcn_s_wait_asynccnt)
  __builtin_amdgcn_s_wait_asynccnt(0);
#else
  asm volatile("s_wait_asynccnt 0" ::: "memory");
#endif
#endif
}

// ---------------------------------------------------------------------------
// fp32 -> bf16 conversion
// ---------------------------------------------------------------------------
__global__ __launch_bounds__(256) void cvt_bf16_kernel(const float* __restrict__ in,
                                                       __bf16* __restrict__ out, int n) {
  int i = blockIdx.x * 256 + threadIdx.x;
  if (i < n) out[i] = f2bf(in[i]);
}

// ---------------------------------------------------------------------------
// bf16 WMMA GEMM:  C[m,n] = sum_k A[m,k] * W[n,k]     (M=4096, K=N=1024)
// 256 threads = 8 waves (2 along M, 4 along N); per-wave 32x32 tile (2x2 frags)
// MODE 0: store bf16 into half-head layout  [b, hh(32), t, 32]
// MODE 2: store bf16 transposed V layout    [b, h(16), d(64), n]
// MODE 3: store fp32 row-major (final output)
// ---------------------------------------------------------------------------
template <int MODE>
__global__ __launch_bounds__(256) void gemm_bf16_kernel(const __bf16* __restrict__ A,
                                                        const __bf16* __restrict__ W,
                                                        void* __restrict__ outv) {
  constexpr int K = 1024, NN = 1024;
  const int lane = threadIdx.x & 31;
  const int w    = threadIdx.x >> 5;
  const int l16  = lane & 15;
  const int hi   = lane >> 4;
  const int m0 = blockIdx.y * 64 + (w >> 2) * 32;
  const int n0 = blockIdx.x * 128 + (w & 3) * 32;

  v8f acc[2][2];
#pragma unroll
  for (int i = 0; i < 2; ++i)
#pragma unroll
    for (int j = 0; j < 2; ++j) acc[i][j] = v8f{};

  for (int k0 = 0; k0 < K; k0 += 32) {
    if (k0 + 32 < K) {  // pull next k-slice toward the caches
      __builtin_prefetch(A + (size_t)(m0 + lane) * K + k0 + 32, 0, 3);
      __builtin_prefetch(W + (size_t)(n0 + lane) * K + k0 + 32, 0, 3);
    }
    v16bf af[2], bf[2];
#pragma unroll
    for (int i = 0; i < 2; ++i)
      af[i] = load_a_frag(A + (size_t)(m0 + i * 16) * K + k0, K, l16, hi);
#pragma unroll
    for (int j = 0; j < 2; ++j)
      bf[j] = load_b_frag(W + (size_t)(n0 + j * 16) * K + k0, K, l16, hi);
#pragma unroll
    for (int i = 0; i < 2; ++i)
#pragma unroll
      for (int j = 0; j < 2; ++j)
        acc[i][j] = __builtin_amdgcn_wmma_f32_16x16x32_bf16(
            false, af[i], false, bf[j], (short)0, acc[i][j], false, false);
  }

#pragma unroll
  for (int i = 0; i < 2; ++i) {
#pragma unroll
    for (int r = 0; r < 8; ++r) {
      const int m = m0 + i * 16 + r + hi * 8;
#pragma unroll
      for (int j = 0; j < 2; ++j) {
        const int n = n0 + j * 16 + l16;
        const float val = acc[i][j][r];
        if (MODE == 3) {
          ((float*)outv)[(size_t)m * NN + n] = val;
        } else if (MODE == 0) {           // Q/K half-head layout
          const int b = m >> 11, t = m & 2047;
          const int hh = n >> 5, d = n & 31;
          ((__bf16*)outv)[(((size_t)(b * 32 + hh) * 2048) + t) * 32 + d] = f2bf(val);
        } else {                          // MODE 2: V transposed [b,h,d,n]
          const int b = m >> 11, np = m & 2047;
          const int h = n >> 6, d = n & 63;
          ((__bf16*)outv)[(((size_t)(b * 16 + h) * 64) + d) * 2048 + np] = f2bf(val);
        }
      }
    }
  }
}

// ---------------------------------------------------------------------------
// Differential flash attention, cooperative K/V staging via async VMEM->LDS.
// grid = B*H*(T/64); block = 256 (8 waves). Waves 0-3: half 0, waves 4-7:
// half 1; wave handles 16 rows. Block-uniform n-loop (to the 64-row tile's
// causal frontier) so all waves share double-buffered K/V LDS tiles.
// Row-sums of P are computed with an extra WMMA against an all-ones matrix
// (row-sum lands in D VGPR r for every lane == exactly the l_r layout).
// ---------------------------------------------------------------------------
__global__ __launch_bounds__(256) void diffattn_kernel(
    const __bf16* __restrict__ Qbf, const __bf16* __restrict__ Kbf,
    const __bf16* __restrict__ Vt,
    const float* __restrict__ lq1, const float* __restrict__ lq2,
    const float* __restrict__ lk1, const float* __restrict__ lk2,
    const float* __restrict__ g, __bf16* __restrict__ Xbuf) {
  __shared__ alignas(16) __bf16 kbuf[2][2][32][32];  // [buf][half][n][d]  8 KB
  __shared__ alignas(16) __bf16 vbuf[2][64][32];     // [buf][d][n]        8 KB
  __shared__ alignas(16) __bf16 pbuf[8][16][32];     // per-wave P staging 8 KB
  __shared__ float obuf[2][64][DH + 1];              // half -> row -> d  ~33 KB
  __shared__ float s_lambda;

  const int tid  = threadIdx.x;
  const int lane = tid & 31;
  const int w    = tid >> 5;
  const int l16  = lane & 15;
  const int hi   = lane >> 4;

  const int ttile = blockIdx.x % (TDIM / 64);
  const int h     = (blockIdx.x / (TDIM / 64)) % NHEAD;
  const int b     = blockIdx.x / ((TDIM / 64) * NHEAD);

  if (tid == 0) {
    float a0 = 0.f, a1 = 0.f;
    for (int i = 0; i < HALF; ++i) { a0 += lq1[i] * lk1[i]; a1 += lq2[i] * lk2[i]; }
    s_lambda = __expf(a0) - __expf(a1) + LAMBDA_INIT;
  }
  __syncthreads();

  const int half = w >> 2;
  const int rt   = w & 3;
  const int t0   = ttile * 64 + rt * 16;

  const __bf16* khead0 = Kbf + (size_t)(b * 2 * NHEAD + 2 * h + 0) * NPOS * HALF;
  const __bf16* khead1 = Kbf + (size_t)(b * 2 * NHEAD + 2 * h + 1) * NPOS * HALF;
  const __bf16* vhead  = Vt + (size_t)(b * NHEAD + h) * DH * NPOS;
  const __bf16* qhead  = Qbf + (size_t)(b * 2 * NHEAD + 2 * h + half) * TDIM * HALF;

  // This thread's staging assignment (whole block cooperates):
  //  K: 2 halves x 32 n-rows x 4 d-chunks of 8  -> 256 x 16B
  //  V: 64 d-rows x 4 n-chunks of 8             -> 256 x 16B
  const int kh = tid >> 7;
  const int kn = (tid & 127) >> 2;
  const int kd = (tid & 3) * 8;
  const int vd = tid >> 2;
  const int vn = (tid & 3) * 8;
  const __bf16* ksrc_base = (kh ? khead1 : khead0) + (size_t)kn * HALF + kd;

  auto stage = [&](int buf, int n0) {
    async_cp16(ksrc_base + (size_t)n0 * HALF, &kbuf[buf][kh][kn][kd]);
    async_cp16(vhead + (size_t)vd * NPOS + n0 + vn, &vbuf[buf][vd][vn]);
  };

  const v16bf qf = load_a_frag(qhead + (size_t)t0 * HALF, HALF, l16, hi);

  v16bf onesf;
#pragma unroll
  for (int e = 0; e < 16; ++e) onesf[e] = f2bf(1.0f);

  float m_r[8], l_r[8];
  v8f o[4];
#pragma unroll
  for (int r = 0; r < 8; ++r) { m_r[r] = -1e30f; l_r[r] = 0.f; }
#pragma unroll
  for (int j = 0; j < 4; ++j) o[j] = v8f{};

  const int nend = t0 + 16;            // this wave's causal frontier
  const int Nblk = ttile * 64 + 64;    // block-uniform frontier

  stage(0, 0);
  for (int n0 = 0; n0 < Nblk; n0 += 32) {
    const int buf = (n0 >> 5) & 1;
    wait_async_lds();
    __syncthreads();                   // staged tiles visible to all waves
    if (n0 + 32 < Nblk) stage(buf ^ 1, n0 + 32);

    if (n0 < nend) {
      // ---- S = Q K^T for two 16-wide column tiles (from LDS)
      v8f s[2];
#pragma unroll
      for (int jt = 0; jt < 2; ++jt) {
        const __bf16* kp = &kbuf[buf][half][jt * 16 + l16][hi * 16];
        v16bf kf = cat16(*reinterpret_cast<const v8bf*>(kp),
                         *reinterpret_cast<const v8bf*>(kp + 8));
        v8f z{};
        s[jt] = __builtin_amdgcn_wmma_f32_16x16x32_bf16(false, qf, false, kf,
                                                        (short)0, z, false, false);
      }
      // ---- scale + causal mask
#pragma unroll
      for (int jt = 0; jt < 2; ++jt) {
        const int n = n0 + jt * 16 + l16;
#pragma unroll
        for (int r = 0; r < 8; ++r) {
          const int t = t0 + r + hi * 8;
          const float v = s[jt][r] * QK_SCALE;
          s[jt][r] = (n <= t) ? v : -1e30f;
        }
      }
      // ---- online softmax: row max via cross-lane reduce, exp, rescale O
      float sc_r[8];
#pragma unroll
      for (int r = 0; r < 8; ++r) {
        float tmax = fmaxf(s[0][r], s[1][r]);
#pragma unroll
        for (int msk = 1; msk < 16; msk <<= 1)
          tmax = fmaxf(tmax, __shfl_xor(tmax, msk, 32));
        const float nm = fmaxf(m_r[r], tmax);
        sc_r[r] = __expf(m_r[r] - nm);
        m_r[r] = nm;
        const float e0 = __expf(s[0][r] - nm);
        const float e1 = __expf(s[1][r] - nm);
#pragma unroll
        for (int j = 0; j < 4; ++j) o[j][r] *= sc_r[r];
        // stash P (C-layout -> row-major LDS tile, bf16)
        pbuf[w][r + hi * 8][l16]      = f2bf(e0);
        pbuf[w][r + hi * 8][16 + l16] = f2bf(e1);
      }
      asm volatile("s_wait_dscnt 0" ::: "memory");  // wave-local LDS RAW
      // ---- reload P as an A fragment
      const v16bf pf = load_a_frag(&pbuf[w][0][0], 32, l16, hi);
      // ---- row sums of P via WMMA against ones: D[r] = rowsum(row r+hi*8)
      {
        v8f z{};
        v8f sums = __builtin_amdgcn_wmma_f32_16x16x32_bf16(false, pf, false, onesf,
                                                           (short)0, z, false, false);
#pragma unroll
        for (int r = 0; r < 8; ++r) l_r[r] = l_r[r] * sc_r[r] + sums[r];
      }
      // ---- O += P * V over 4 d-tiles (V tile in LDS, [d][n])
#pragma unroll
      for (int j = 0; j < 4; ++j) {
        const __bf16* vp = &vbuf[buf][j * 16 + l16][hi * 16];
        v16bf vf = cat16(*reinterpret_cast<const v8bf*>(vp),
                         *reinterpret_cast<const v8bf*>(vp + 8));
        o[j] = __builtin_amdgcn_wmma_f32_16x16x32_bf16(false, pf, false, vf,
                                                       (short)0, o[j], false, false);
      }
    }
  }

  // ---- write normalized per-half outputs to shared
#pragma unroll
  for (int r = 0; r < 8; ++r) {
    const float rinv = __builtin_amdgcn_rcpf(l_r[r]);
#pragma unroll
    for (int j = 0; j < 4; ++j)
      obuf[half][rt * 16 + r + hi * 8][j * 16 + l16] = o[j][r] * rinv;
  }
  __syncthreads();

  // ---- combine halves, RMSNorm over DH, store bf16 for final GEMM
  if (tid < 64) {
    const float lam = s_lambda;
    float vrow[DH];
    float ss = 0.f;
#pragma unroll
    for (int d = 0; d < DH; ++d) {
      const float v = obuf[0][tid][d] - lam * obuf[1][tid][d];
      vrow[d] = v;
      ss += v * v;
    }
    const float rinv = rsqrtf(ss * (1.0f / DH) + 1e-5f) * (1.0f - LAMBDA_INIT);
    const int t = ttile * 64 + tid;
    __bf16* outp = Xbuf + ((size_t)(b * TDIM + t) * HID) + h * DH;
#pragma unroll
    for (int d = 0; d < DH; ++d) outp[d] = f2bf(vrow[d] * rinv * g[d]);
  }
}

// ---------------------------------------------------------------------------
// Host-side orchestration
// ---------------------------------------------------------------------------
extern "C" void kernel_launch(void* const* d_in, const int* in_sizes, int n_in,
                              void* d_out, int out_size, void* d_ws, size_t ws_size,
                              hipStream_t stream) {
  (void)in_sizes; (void)n_in; (void)out_size; (void)ws_size;
  const float* query = (const float*)d_in[0];
  const float* key_  = (const float*)d_in[1];
  const float* value = (const float*)d_in[2];
  const float* Wq    = (const float*)d_in[3];
  const float* Wk    = (const float*)d_in[4];
  const float* Wv    = (const float*)d_in[5];
  const float* Wo    = (const float*)d_in[6];
  const float* lq1   = (const float*)d_in[7];
  const float* lq2   = (const float*)d_in[8];
  const float* lk1   = (const float*)d_in[9];
  const float* lk2   = (const float*)d_in[10];
  const float* g     = (const float*)d_in[11];
  float* out = (float*)d_out;

  const int M = BBATCH * TDIM;            // 4096
  const size_t actN = (size_t)M * HID;    // 4 Mi elems
  const size_t wN   = (size_t)HID * HID;  // 1 Mi elems

  char* ws = (char*)d_ws;
  size_t off = 0;
  auto carve = [&](size_t elems) {
    __bf16* p = (__bf16*)(ws + off);
    off += ((elems * sizeof(__bf16)) + 255) & ~(size_t)255;
    return p;
  };
  __bf16* bq_in = carve(actN);
  __bf16* bk_in = carve(actN);
  __bf16* bv_in = carve(actN);
  __bf16* bWq   = carve(wN);
  __bf16* bWk   = carve(wN);
  __bf16* bWv   = carve(wN);
  __bf16* bWo   = carve(wN);
  __bf16* Qbf   = carve(actN);            // [b, hh, t, 32]
  __bf16* Kbf   = carve(actN);            // [b, hh, n, 32]
  __bf16* Vt    = carve(actN);            // [b, h, d, n]
  __bf16* Xbuf  = carve(actN);            // [b, t, hid]

  // 1) fp32 -> bf16
  auto cvt = [&](const float* src, __bf16* dst, size_t n) {
    cvt_bf16_kernel<<<dim3((unsigned)((n + 255) / 256)), dim3(256), 0, stream>>>(src, dst, (int)n);
  };
  cvt(query, bq_in, actN);
  cvt(key_,  bk_in, actN);
  cvt(value, bv_in, actN);
  cvt(Wq, bWq, wN);
  cvt(Wk, bWk, wN);
  cvt(Wv, bWv, wN);
  cvt(Wo, bWo, wN);

  // 2) projections (WMMA GEMMs)
  dim3 ggrid(HID / 128, M / 64), gblk(256);
  gemm_bf16_kernel<0><<<ggrid, gblk, 0, stream>>>(bq_in, bWq, Qbf);
  gemm_bf16_kernel<0><<<ggrid, gblk, 0, stream>>>(bk_in, bWk, Kbf);
  gemm_bf16_kernel<2><<<ggrid, gblk, 0, stream>>>(bv_in, bWv, Vt);

  // 3) differential flash attention + RMSNorm
  diffattn_kernel<<<dim3(BBATCH * NHEAD * (TDIM / 64)), dim3(256), 0, stream>>>(
      Qbf, Kbf, Vt, lq1, lq2, lk1, lk2, g, Xbuf);

  // 4) output projection (fp32 store)
  gemm_bf16_kernel<3><<<ggrid, gblk, 0, stream>>>(Xbuf, bWo, out);
}
